// CycleFC_7662221656719
// MI455X (gfx1250) — matrix operand order repaired and checked by reading, not compiled, not run on hardware
//
#include <hip/hip_runtime.h>
#include <stdint.h>

typedef __attribute__((ext_vector_type(16))) __bf16 v16bf;
typedef __attribute__((ext_vector_type(8)))  float  v8f;

#define CIN   384
#define COUT  384
#define BATCH 16
#define HH    64
#define WW    64
#define K_DIM (BATCH * CIN)   // 6144
#define N_DIM (HH * WW)       // 4096
#define M_DIM COUT            // 384

// float -> bf16 (round-to-nearest-even), no header dependence
__device__ __forceinline__ uint16_t f2bf(float f) {
    uint32_t u = __builtin_bit_cast(uint32_t, f);
    uint32_t r = (u + 0x7FFFu + ((u >> 16) & 1u)) >> 16;
    return (uint16_t)r;
}

// A[m][k] = bf16( W_mlp[b][c][m] ),  k = b*CIN + c.  Row-major, K contiguous.
// Each thread emits 4 consecutive k (one 8-byte store).
__global__ void convert_w_kernel(const float* __restrict__ Wm,
                                 uint16_t* __restrict__ A) {
    int tid = blockIdx.x * blockDim.x + threadIdx.x;
    if (tid >= M_DIM * K_DIM / 4) return;
    int m  = tid / (K_DIM / 4);
    int k4 = (tid - m * (K_DIM / 4)) * 4;
    int b  = k4 / CIN;
    int c  = k4 - b * CIN;          // CIN % 4 == 0 -> c..c+3 same b
    const float* src = Wm + ((long)b * CIN + c) * COUT + m;
    ushort4 o;
    o.x = f2bf(src[0 * COUT]);
    o.y = f2bf(src[1 * COUT]);
    o.z = f2bf(src[2 * COUT]);
    o.w = f2bf(src[3 * COUT]);
    *(ushort4*)(A + (long)m * K_DIM + k4) = o;
}

// BT[n][k] = bf16( x[b][c][(h+di)%H][(w+dj)%W] ),  n = h*WW + w, k = b*CIN + c.
// The CycleFC cyclic shift is folded in here so the GEMM is dense.
// Each thread emits 4 consecutive k (one 8-byte store).
__global__ void convert_x_kernel(const float* __restrict__ x,
                                 uint16_t* __restrict__ BT) {
    long tid = (long)blockIdx.x * blockDim.x + threadIdx.x;
    if (tid >= (long)N_DIM * K_DIM / 4) return;
    int k4 = (int)(tid % (K_DIM / 4)) * 4;
    int n  = (int)(tid / (K_DIM / 4));
    int h = n / WW, w = n - (n / WW) * WW;
    int b = k4 / CIN;
    int c = k4 - b * CIN;           // c..c+3 same b
    ushort4 o;
    {
        int di = c % 3 - 1, dj = (c / 3) % 3 - 1;
        int ii = (h + di + HH) % HH, jj = (w + dj + WW) % WW;
        o.x = f2bf(x[(((long)b * CIN + c) * HH + ii) * WW + jj]);
    }
    {
        int cc = c + 1;
        int di = cc % 3 - 1, dj = (cc / 3) % 3 - 1;
        int ii = (h + di + HH) % HH, jj = (w + dj + WW) % WW;
        o.y = f2bf(x[(((long)b * CIN + cc) * HH + ii) * WW + jj]);
    }
    {
        int cc = c + 2;
        int di = cc % 3 - 1, dj = (cc / 3) % 3 - 1;
        int ii = (h + di + HH) % HH, jj = (w + dj + WW) % WW;
        o.z = f2bf(x[(((long)b * CIN + cc) * HH + ii) * WW + jj]);
    }
    {
        int cc = c + 3;
        int di = cc % 3 - 1, dj = (cc / 3) % 3 - 1;
        int ii = (h + di + HH) % HH, jj = (w + dj + WW) % WW;
        o.w = f2bf(x[(((long)b * CIN + cc) * HH + ii) * WW + jj]);
    }
    *(ushort4*)(BT + (long)n * K_DIM + k4) = o;
}

// GEMM: out[m][n] = sum_k A[m][k]*BT[n][k] + bias[m], written to batch slice 0.
// Block = 256 threads = 8 waves arranged 2(M) x 4(N); each wave owns a 32x64
// output tile as a 2x4 grid of 16x16 WMMA accumulators.
// Per K-step: 6 fragment loads (12 x b128) feed 8 WMMAs -> 1.5 loads/WMMA,
// 64 FLOP per L2 byte on A, 32 on B; both operands are L2-resident.
__global__ __launch_bounds__(256)
void wmma_gemm_kernel(const uint16_t* __restrict__ A,
                      const uint16_t* __restrict__ BT,
                      const float* __restrict__ bias,
                      float* __restrict__ out) {
    const int lane  = threadIdx.x & 31;
    const int wid   = threadIdx.x >> 5;
    const int waveM = wid & 1;          // 2 waves in M
    const int waveN = wid >> 1;         // 4 waves in N
    const int m0 = blockIdx.y * 64  + waveM * 32;   // wave: rows m0..m0+31
    const int n0 = blockIdx.x * 256 + waveN * 64;   // wave: cols n0..n0+63
    const int hi  = lane >> 4;          // half-wave select
    const int l15 = lane & 15;

    // A fragment (16x32 bf16): lane<16 holds row m+l15, K 0..15;
    // lane>=16 holds the same row's K 16..31 (matches B fragment K order).
    const uint16_t* aptr[2];
#pragma unroll
    for (int i = 0; i < 2; ++i)
        aptr[i] = A + (long)(m0 + i * 16 + l15) * K_DIM + hi * 16;

    // B fragment (32x16 bf16): lane<16 holds col n+l15, K 0..15 packed 2/VGPR;
    // lane>=16 holds K 16..31 of that col. BT row-major => contiguous 32B/lane.
    const uint16_t* bptr[4];
#pragma unroll
    for (int j = 0; j < 4; ++j)
        bptr[j] = BT + (long)(n0 + j * 16 + l15) * K_DIM + hi * 16;

    // Accumulators pre-loaded with bias: VGPR r holds M = m + r + hi*8 (C layout).
    v8f acc[2][4];
#pragma unroll
    for (int i = 0; i < 2; ++i) {
#pragma unroll
        for (int r = 0; r < 8; ++r) {
            float bv = bias[m0 + i * 16 + r + hi * 8];
#pragma unroll
            for (int j = 0; j < 4; ++j) acc[i][j][r] = bv;
        }
    }

    typedef union { uint32_t u[8]; v16bf v; } frag_t;

    for (int k = 0; k < K_DIM; k += 32) {
        frag_t af[2], bf[4];
#pragma unroll
        for (int i = 0; i < 2; ++i) {
            const uint4 q0 = *(const uint4*)(aptr[i] + k);
            const uint4 q1 = *(const uint4*)(aptr[i] + k + 8);
            af[i].u[0]=q0.x; af[i].u[1]=q0.y; af[i].u[2]=q0.z; af[i].u[3]=q0.w;
            af[i].u[4]=q1.x; af[i].u[5]=q1.y; af[i].u[6]=q1.z; af[i].u[7]=q1.w;
        }
#pragma unroll
        for (int j = 0; j < 4; ++j) {
            const uint4 q0 = *(const uint4*)(bptr[j] + k);
            const uint4 q1 = *(const uint4*)(bptr[j] + k + 8);
            bf[j].u[0]=q0.x; bf[j].u[1]=q0.y; bf[j].u[2]=q0.z; bf[j].u[3]=q0.w;
            bf[j].u[4]=q1.x; bf[j].u[5]=q1.y; bf[j].u[6]=q1.z; bf[j].u[7]=q1.w;
        }
#pragma unroll
        for (int i = 0; i < 2; ++i) {
#pragma unroll
            for (int j = 0; j < 4; ++j) {
                acc[i][j] = __builtin_amdgcn_wmma_f32_16x16x32_bf16(
                    false, af[i].v, false, bf[j].v,
                    (short)0, acc[i][j], false, false);
            }
        }
    }

    // C/D layout: VGPR r, lanes 0-15 -> M = r, lanes 16-31 -> M = r + 8; N = lane%16.
#pragma unroll
    for (int i = 0; i < 2; ++i) {
#pragma unroll
        for (int r = 0; r < 8; ++r) {
            const long row = (long)(m0 + i * 16 + r + hi * 8) * N_DIM;
#pragma unroll
            for (int j = 0; j < 4; ++j) {
                out[row + n0 + j * 16 + l15] = acc[i][j][r];
            }
        }
    }
}

// Broadcast batch slice 0 into slices 1..15 (float4, reads hit L2).
__global__ void broadcast_kernel(float* __restrict__ out) {
    long tid = (long)blockIdx.x * blockDim.x + threadIdx.x;
    const long slice4 = (long)M_DIM * N_DIM / 4;  // 393216 float4s
    if (tid >= slice4) return;
    float4 v = ((const float4*)out)[tid];
#pragma unroll
    for (int b = 1; b < BATCH; ++b) {
        ((float4*)out)[b * slice4 + tid] = v;
    }
}

extern "C" void kernel_launch(void* const* d_in, const int* in_sizes, int n_in,
                              void* d_out, int out_size, void* d_ws, size_t ws_size,
                              hipStream_t stream) {
    const float* x    = (const float*)d_in[0];   // (16, 384, 64, 64)
    const float* Wm   = (const float*)d_in[1];   // (16, 384, 384)
    const float* bias = (const float*)d_in[2];   // (384,)
    float* out = (float*)d_out;                  // (16, 384, 64, 64)

    uint16_t* BT = (uint16_t*)d_ws;                    // 4096 x 6144 bf16 (~50 MB)
    uint16_t* A  = BT + (size_t)N_DIM * K_DIM;         // 384  x 6144 bf16 (~4.7 MB)

    {
        long total = (long)N_DIM * K_DIM / 4;          // 6,291,456
        int blocks = (int)((total + 255) / 256);
        convert_x_kernel<<<blocks, 256, 0, stream>>>(x, BT);
    }
    {
        int total = M_DIM * K_DIM / 4;                 // 589,824
        int blocks = (total + 255) / 256;
        convert_w_kernel<<<blocks, 256, 0, stream>>>(Wm, A);
    }
    {
        dim3 grid(N_DIM / 256, M_DIM / 64);            // 16 x 6 = 96 blocks
        wmma_gemm_kernel<<<grid, 256, 0, stream>>>(A, BT, bias, out);
    }
    {
        long total = (long)M_DIM * N_DIM / 4;
        int blocks = (int)((total + 255) / 256);
        broadcast_kernel<<<blocks, 256, 0, stream>>>(out);
    }
}
